// FourLayer_64F_86663850099092
// MI455X (gfx1250) — compile-verified, matrix operand order checked by reading, not run on hardware
//
#include <hip/hip_runtime.h>
#include <hip/hip_bf16.h>

// ---------------------------------------------------------------------------
// CDNA5 (gfx1250) DN4-style few-shot forward pass.
// Every GEMM-shaped op (conv1..conv4, cw1, cw2, cosine einsum) runs on
// v_wmma_f32_16x16x32_f16.  Operands are pre-swizzled into WMMA fragment
// order (each fragment = contiguous 32B -> 2 x ds_load_b128) and bulk
// global->LDS staging uses the Tensor Data Mover where available.
// ---------------------------------------------------------------------------

typedef __attribute__((ext_vector_type(16))) _Float16 v16h;
typedef __attribute__((ext_vector_type(8)))  _Float16 v8h;
typedef __attribute__((ext_vector_type(8)))  float    v8f;
typedef __attribute__((ext_vector_type(4)))  unsigned int u32x4;
typedef __attribute__((ext_vector_type(8)))  int          i32x8;
typedef __attribute__((ext_vector_type(4)))  int          i32x4;

#define DEVINL __device__ __forceinline__

#if defined(__has_builtin)
#if __has_builtin(__builtin_amdgcn_tensor_load_to_lds) && \
    __has_builtin(__builtin_amdgcn_s_wait_tensorcnt)
#define HAVE_TDM 1
#endif
#endif
#ifndef HAVE_TDM
#define HAVE_TDM 0
#endif

DEVINL v8f wmma16(v16h a, v16h b, v8f c) {
  // (neg_a, A, neg_b, B, c_mod, C, reuse_a, reuse_b)
  return __builtin_amdgcn_wmma_f32_16x16x32_f16(false, a, false, b, (short)0, c,
                                                false, false);
}

// ISA 7.12.2 16-bit A/B fragment: lane l holds row/col (l&15),
// K-base = 8*(l>>4); element j maps to K = kbase + j (+8 for j>=8).
DEVINL int frag_k(int kbase, int j) { return kbase + j + ((j >= 8) ? 8 : 0); }

#if HAVE_TDM
// 1-D TDM descriptor: copy nelem f16 (nelem <= 65535) global -> LDS.
DEVINL void tdm_load_1d(unsigned lds_off, const void* gptr, unsigned nelem) {
  unsigned long long ga = (unsigned long long)(size_t)gptr;
  u32x4 g0;
  g0[0] = 1u;                                    // count=1 (user descriptor)
  g0[1] = lds_off;                               // lds_addr
  g0[2] = (unsigned)(ga & 0xFFFFFFFFu);          // global_addr[31:0]
  g0[3] = (unsigned)((ga >> 32) & 0x01FFFFFFu)   // global_addr[56:32]
          | (2u << 30);                          // type=2 ("image")
  i32x8 g1;
  g1[0] = 0x00010000;                            // wg_mask=0, data_size=2B
  g1[1] = (int)((nelem & 0xFFFFu) << 16);        // tensor_dim0 lo16
  g1[2] = (int)((nelem >> 16) & 0xFFFFu) | (1 << 16); // dim0 hi16 | tensor_dim1=1
  g1[3] = (int)((nelem & 0xFFFFu) << 16);        // tile_dim0
  g1[4] = 0;                                     // tile_dim1=0, tile_dim2=0
  g1[5] = (int)nelem;                            // tensor_dim0_stride lo32
  g1[6] = 0;
  g1[7] = 0;
  i32x4 z4 = {0, 0, 0, 0};
#if __clang_major__ >= 23
  i32x8 z8 = {0, 0, 0, 0, 0, 0, 0, 0};
  __builtin_amdgcn_tensor_load_to_lds(g0, g1, z4, z4, z8, 0);
#else
  __builtin_amdgcn_tensor_load_to_lds(g0, g1, z4, z4, 0);
#endif
}
#endif

DEVINL float lrelu(float v) { return v >= 0.f ? v : 0.2f * v; }

// ---------------------------------------------------------------------------
// Weight pre-swizzle: w (Cout=64, Kvalid) f32 -> fragment-ordered f16 panel,
// layout [wave(4)][kb(KB)][lane(32)][j(16)], zero-padded for K >= Kvalid.
// ---------------------------------------------------------------------------
__global__ void prep_wfrag_kernel(const float* __restrict__ w,
                                  _Float16* __restrict__ outp,
                                  int KB, int Kvalid) {
  int f = blockIdx.x * blockDim.x + threadIdx.x;
  if (f >= 4 * KB * 32) return;
  int lane = f & 31, g = f >> 5;
  int kb = g % KB, wave = g / KB;
  int m = wave * 16 + (lane & 15);
  int kbase = kb * 32 + ((lane >> 4) << 3);
  _Float16 tmp[16];
#pragma unroll
  for (int j = 0; j < 16; ++j) {
    int k = frag_k(kbase, j);
    tmp[j] = (k < Kvalid) ? (_Float16)w[m * Kvalid + k] : (_Float16)0.f;
  }
  v8h lo, hi;
#pragma unroll
  for (int j = 0; j < 8; ++j) { lo[j] = tmp[j]; hi[j] = tmp[j + 8]; }
  v8h* o = (v8h*)(outp + (size_t)f * 16);
  o[0] = lo; o[1] = hi;
}

// ---------------------------------------------------------------------------
// Generic Cin->64 3x3 conv as WMMA GEMM.  M=64(Cout) x K=KB*32 x N=64
// positions per workgroup (128 threads / 4 waves).  Weights fragment-ordered
// via TDM; im2col tile built directly in fragment order
// [kb(KB)][nt(4)][lane(32)][j(16)]; K >= Cin*9 is zero padding.
// ---------------------------------------------------------------------------
__global__ void conv_wmma_kernel(const _Float16* __restrict__ x,
                                 const _Float16* __restrict__ wfrag,
                                 _Float16* __restrict__ out,
                                 int Hin, int Win, int Hout, int Wout,
                                 int stride, int pad, int KB, int Cin) {
  extern __shared__ char smem_raw[];
  _Float16* Aw = (_Float16*)smem_raw;             // 4*KB*512 (frag order)
  _Float16* Bt = Aw + (size_t)4 * KB * 512;       // KB*4*512 (frag order)
  const int b     = blockIdx.y;
  const int tile0 = blockIdx.x * 64;
  const int Npos  = Hout * Wout;
  const int tid   = threadIdx.x;
  const int HWin  = Hin * Win;
  const int Kvalid = Cin * 9;
  const _Float16* xb = x + (size_t)b * Cin * HWin;
  const int wave = tid >> 5, lane = tid & 31;

  __builtin_prefetch(xb + tid * 64, 0, 1);        // global_prefetch_b8

#if HAVE_TDM
  if (wave == 0) {
    tdm_load_1d((unsigned)(size_t)Aw, wfrag, (unsigned)(4 * KB * 512));
    __builtin_amdgcn_s_wait_tensorcnt(0);
  }
#else
  for (int i = tid; i < 4 * KB * 512 / 8; i += 128)
    ((v8h*)Aw)[i] = ((const v8h*)wfrag)[i];
#endif

  // im2col into fragment order: one thread per 16-element fragment.
  for (int f = tid; f < KB * 4 * 32; f += 128) {
    int flane = f & 31, g = f >> 5;
    int nt = g & 3, kb = g >> 2;
    int n = nt * 16 + (flane & 15);
    int pos = tile0 + n;
    int oy = pos / Wout, ox = pos - oy * Wout;
    int kbase = kb * 32 + ((flane >> 4) << 3);
    _Float16 tmp[16];
#pragma unroll
    for (int j = 0; j < 16; ++j) {
      int k = frag_k(kbase, j);
      _Float16 v = (_Float16)0.f;
      if (k < Kvalid) {
        int ci = k / 9, r = k - ci * 9;
        int kh = r / 3, kw = r - kh * 3;
        int iy = oy * stride - pad + kh;
        int ix = ox * stride - pad + kw;
        if (pos < Npos && iy >= 0 && iy < Hin && ix >= 0 && ix < Win)
          v = xb[ci * HWin + iy * Win + ix];
      }
      tmp[j] = v;
    }
    v8h lo, hi;
#pragma unroll
    for (int j = 0; j < 8; ++j) { lo[j] = tmp[j]; hi[j] = tmp[j + 8]; }
    v8h* o = (v8h*)(Bt + (size_t)f * 16);
    o[0] = lo; o[1] = hi;
  }
  __syncthreads();

  const int hi_ = lane >> 4, col = lane & 15;
  v8f c0 = {}, c1 = {}, c2 = {}, c3 = {};
  const _Float16* Af = Aw + (size_t)wave * KB * 512;
  for (int kb = 0; kb < KB; ++kb) {
    v16h a = *(const v16h*)(Af + (size_t)kb * 512 + lane * 16);
    const _Float16* Bf = Bt + (size_t)kb * 4 * 512;
    c0 = wmma16(a, *(const v16h*)(Bf + 0 * 512 + lane * 16), c0);
    c1 = wmma16(a, *(const v16h*)(Bf + 1 * 512 + lane * 16), c1);
    c2 = wmma16(a, *(const v16h*)(Bf + 2 * 512 + lane * 16), c2);
    c3 = wmma16(a, *(const v16h*)(Bf + 3 * 512 + lane * 16), c3);
  }
  v8f cs[4] = {c0, c1, c2, c3};
#pragma unroll
  for (int t = 0; t < 4; ++t) {
#pragma unroll
    for (int r = 0; r < 8; ++r) {
      int co  = wave * 16 + hi_ * 8 + r;  // D: M = vgpr + 8*(lane>=16)
      int pos = tile0 + t * 16 + col;     // N = lane&15
      if (pos < Npos)
        out[((size_t)b * 64 + co) * Npos + pos] = (_Float16)cs[t][r];
    }
  }
}

// BN batch statistics, one channel per block (deterministic tree reduce).
__global__ void bn_stats_kernel(const _Float16* __restrict__ raw,
                                const float* __restrict__ g,
                                const float* __restrict__ bb,
                                float* __restrict__ scsh, int B, int HW) {
  __shared__ float s1[256], s2[256];
  int c = blockIdx.x, tid = threadIdx.x;
  float sum = 0.f, sq = 0.f;
  int N = B * HW;
  for (int j = tid; j < N; j += 256) {
    int b = j / HW, i = j - b * HW;
    float v = (float)raw[((size_t)b * 64 + c) * HW + i];
    sum += v; sq += v * v;
  }
  s1[tid] = sum; s2[tid] = sq; __syncthreads();
  for (int st = 128; st > 0; st >>= 1) {
    if (tid < st) { s1[tid] += s1[tid + st]; s2[tid] += s2[tid + st]; }
    __syncthreads();
  }
  if (tid == 0) {
    float m = s1[0] / N, var = s2[0] / N - m * m;
    float sc = g[c] * rsqrtf(var + 1e-5f);
    scsh[2 * c] = sc; scsh[2 * c + 1] = bb[c] - m * sc;
  }
}

// Fused BN-apply + LeakyReLU + optional pk x pk maxpool.
__global__ void bn_lrelu_pool_kernel(const _Float16* __restrict__ raw,
                                     const float* __restrict__ scsh,
                                     _Float16* __restrict__ out,
                                     int B, int Hi, int Wi, int pk) {
  int Ho = Hi / pk, Wo = Wi / pk;
  int idx = blockIdx.x * blockDim.x + threadIdx.x;
  int total = B * 64 * Ho * Wo;
  if (idx >= total) return;
  int wo = idx % Wo; int t = idx / Wo; int ho = t % Ho; t /= Ho;
  int c = t & 63; int b = t >> 6;
  float sc = scsh[2 * c], sh = scsh[2 * c + 1];
  float m = -1e30f;
  for (int dy = 0; dy < pk; ++dy)
    for (int dx = 0; dx < pk; ++dx) {
      float v = (float)raw[(((size_t)b * 64 + c) * Hi + ho * pk + dy) * Wi +
                           wo * pk + dx] * sc + sh;
      m = fmaxf(m, lrelu(v));
    }
  out[idx] = (_Float16)m;
}

// cw3: 3x3 conv on 3x3 input (pad 0) -> (B,64) + bias + lrelu.
__global__ void cw3_kernel(const _Float16* __restrict__ x,
                           const float* __restrict__ w,
                           const float* __restrict__ bias,
                           float* __restrict__ v, int B) {
  int idx = blockIdx.x * blockDim.x + threadIdx.x;
  if (idx >= B * 64) return;
  int co = idx & 63, b = idx >> 6;
  float acc = bias[co];
  for (int ci = 0; ci < 64; ++ci)
#pragma unroll
    for (int r = 0; r < 9; ++r)
      acc += w[(co * 64 + ci) * 9 + r] * (float)x[((size_t)b * 64 + ci) * 9 + r];
  v[idx] = lrelu(acc);
}

// Generic transposed conv (tiny spatial sizes; scalar is fine).
__global__ void deconv_kernel(const _Float16* __restrict__ in,
                              const float* __restrict__ w,
                              const float* __restrict__ bias,
                              _Float16* __restrict__ out_h,
                              float* __restrict__ out_f,
                              int B, int Cin, int Cout, int K, int S,
                              int Hin, int Hout) {
  int idx = blockIdx.x * blockDim.x + threadIdx.x;
  int total = B * Cout * Hout * Hout;
  if (idx >= total) return;
  int xx_ = idx % Hout; int t = idx / Hout; int y = t % Hout; t /= Hout;
  int o = t % Cout; int b = t / Cout;
  float acc = bias ? bias[o] : 0.f;
  for (int kh = 0; kh < K; ++kh) {
    int yy = y - kh; if (yy < 0 || (yy % S)) continue;
    int i = yy / S; if (i >= Hin) continue;
    for (int kw = 0; kw < K; ++kw) {
      int xr = xx_ - kw; if (xr < 0 || (xr % S)) continue;
      int j = xr / S; if (j >= Hin) continue;
      for (int ci = 0; ci < Cin; ++ci)
        acc += w[((ci * Cout + o) * K + kh) * K + kw] *
               (float)in[(((size_t)b * Cin + ci) * Hin + i) * Hin + j];
    }
  }
  if (out_h) out_h[idx] = (_Float16)acc; else out_f[idx] = acc;
}

// argmax over 441 logits -> mu (softmax is monotone).
__global__ void mi_kernel(const float* __restrict__ logits,
                          float* __restrict__ mi, int B) {
  int b = blockIdx.x * blockDim.x + threadIdx.x;
  if (b >= B) return;
  const float* l = logits + b * 441;
  int best = 0; float bv = l[0];
  for (int i = 1; i < 441; ++i) { float v = l[i]; if (v > bv) { bv = v; best = i; } }
  mi[b * 2 + 0] = (float)(best / 21 - 10) * 0.1f;
  mi[b * 2 + 1] = (float)(best % 21 - 10) * 0.1f;
}

// sx/sy/theta heads -> inverse covariance and det.
__global__ void sigma_kernel(const float* __restrict__ v,
                             const float* sxw, const float* sxb,
                             const float* syw, const float* syb,
                             const float* tw,  const float* tb,
                             float* __restrict__ ginv, int B) {
  int b = blockIdx.x * blockDim.x + threadIdx.x;
  if (b >= B) return;
  const float* vb = v + b * 64;
  int ax = 0, ay = 0, at = 0; float bx = -1e30f, by = -1e30f, bt = -1e30f;
  for (int j = 0; j < 6; ++j) {
    float dx = sxb[j], dy = syb[j], dt = tb[j];
    for (int c = 0; c < 64; ++c) {
      float vc = vb[c];
      dx += vc * sxw[j * 64 + c];
      dy += vc * syw[j * 64 + c];
      dt += vc * tw[j * 64 + c];
    }
    if (dx > bx) { bx = dx; ax = j; }
    if (dy > by) { by = dy; ay = j; }
    if (dt > bt) { bt = dt; at = j; }
  }
  const float PI = 3.14159265358979323846f;
  float s1 = (ax / 5.0f) * 0.25f + 0.1f;
  float s2 = (ay / 5.0f) * 0.25f + 0.1f;
  float th = at * (PI / 6.0f);
  float c = __cosf(th), s = __sinf(th);
  float a  = c * c * s1 + s * s * s2;
  float d  = s * s * s1 + c * c * s2;
  float bo = c * s * (s1 - s2);
  float det = a * d - bo * bo;
  ginv[b * 4 + 0] = d / det;  ginv[b * 4 + 1] = -bo / det;
  ginv[b * 4 + 2] = a / det;  ginv[b * 4 + 3] = det;
}

// Gaussian mask over the 21x21 grid.
__global__ void mask_kernel(const float* __restrict__ mi,
                            const float* __restrict__ ginv,
                            float* __restrict__ m, int B) {
  int idx = blockIdx.x * blockDim.x + threadIdx.x;
  if (idx >= B * 441) return;
  int p = idx % 441, b = idx / 441;
  int i = p / 21, j = p % 21;
  float v0 = (i - 10) * 0.1f - mi[b * 2 + 0];
  float v1 = (j - 10) * 0.1f - mi[b * 2 + 1];
  float q = v0 * v0 * ginv[b * 4 + 0] + 2.f * v0 * v1 * ginv[b * 4 + 1] +
            v1 * v1 * ginv[b * 4 + 2];
  const float INV2PI = 0.15915494309189533577f;
  m[idx] = __expf(-0.5f * q) * INV2PI * rsqrtf(ginv[b * 4 + 3]);
}

// qm[b,p] = mask[b,p] / ||feat[b,:,p]||
__global__ void normmask_kernel(const _Float16* __restrict__ feat,
                                const float* __restrict__ m,
                                float* __restrict__ qm, int B) {
  int idx = blockIdx.x * blockDim.x + threadIdx.x;
  if (idx >= B * 441) return;
  int p = idx % 441, b = idx / 441;
  float s = 0.f;
  for (int c = 0; c < 64; ++c) {
    float v = (float)feat[((size_t)b * 64 + c) * 441 + p];
    s += v * v;
  }
  qm[idx] = m[idx] * rsqrtf(s);
}

// ---------------------------------------------------------------------------
// Fragment pre-swizzle for the cosine GEMM.
// A (query): [pass(7)][wave(4)][kb(2)][lane(32)][j(16)], transposed gather.
// B (support): [qt(28)][kb(2)][lane(32)][j(16)].
// ---------------------------------------------------------------------------
__global__ void prep_afrag_kernel(const _Float16* __restrict__ feat,
                                  _Float16* __restrict__ outp, int B) {
  int f = blockIdx.x * blockDim.x + threadIdx.x;
  int nfrag = B * 7 * 4 * 2 * 32;
  if (f >= nfrag) return;
  int lane = f & 31, g = f >> 5;
  int kb = g & 1; g >>= 1;
  int wave = g & 3; g >>= 2;
  int pass = g % 7; int b = g / 7;
  int p = pass * 64 + wave * 16 + (lane & 15);
  int kbase = kb * 32 + ((lane >> 4) << 3);
  _Float16 tmp[16];
#pragma unroll
  for (int j = 0; j < 16; ++j) {
    int c = frag_k(kbase, j);
    tmp[j] = (p < 441) ? feat[((size_t)b * 64 + c) * 441 + p] : (_Float16)0.f;
  }
  v8h lo, hi;
#pragma unroll
  for (int j = 0; j < 8; ++j) { lo[j] = tmp[j]; hi[j] = tmp[j + 8]; }
  v8h* o = (v8h*)(outp + (size_t)f * 16);
  o[0] = lo; o[1] = hi;
}

__global__ void prep_bfrag_kernel(const _Float16* __restrict__ feat,
                                  _Float16* __restrict__ outp, int B) {
  int f = blockIdx.x * blockDim.x + threadIdx.x;
  int nfrag = B * 28 * 2 * 32;
  if (f >= nfrag) return;
  int lane = f & 31, g = f >> 5;
  int kb = g & 1; g >>= 1;
  int qt = g % 28; int b = g / 28;
  int q = qt * 16 + (lane & 15);
  int kbase = kb * 32 + ((lane >> 4) << 3);
  _Float16 tmp[16];
#pragma unroll
  for (int j = 0; j < 16; ++j) {
    int c = frag_k(kbase, j);
    tmp[j] = (q < 441) ? feat[((size_t)b * 64 + c) * 441 + q] : (_Float16)0.f;
  }
  v8h lo, hi;
#pragma unroll
  for (int j = 0; j < 8; ++j) { lo[j] = tmp[j]; hi[j] = tmp[j + 8]; }
  v8h* o = (v8h*)(outp + (size_t)f * 16);
  o[0] = lo; o[1] = hi;
}

// ---------------------------------------------------------------------------
// Cosine einsum + mask + top-3 + sum, fused.  One WG per (a,b): both panels
// (57KB + 57KB, fragment order) TDM-loaded into LDS once; the 73M-element
// 'masked' tensor never exists in memory.  Deterministic LDS reductions.
// ---------------------------------------------------------------------------
__global__ void cossim_topk_kernel(const _Float16* __restrict__ afrag,
                                   const _Float16* __restrict__ bfrag,
                                   const float* __restrict__ qm,
                                   const float* __restrict__ sm,
                                   float* __restrict__ sim) {
  extern __shared__ char smem_raw[];
  _Float16* Af  = (_Float16*)smem_raw;                      // 28672 f16
  _Float16* Bf  = Af + 28672;                               // 28672 f16
  float*    red = (float*)(smem_raw + 114688);              // 64*16*3 f32
  float*    part= (float*)(smem_raw + 114688 + 12288);      // 64 f32
  const int a = blockIdx.x / 5, b = blockIdx.x % 5;
  const int tid = threadIdx.x, wave = tid >> 5, lane = tid & 31;
  const int hi_ = lane >> 4, col = lane & 15;

#if HAVE_TDM
  if (wave == 0) {
    tdm_load_1d((unsigned)(size_t)Af, afrag + (size_t)a * 28672, 28672);
    tdm_load_1d((unsigned)(size_t)Bf, bfrag + (size_t)b * 28672, 28672);
    __builtin_amdgcn_s_wait_tensorcnt(0);
  }
#else
  {
    const v8h* ga = (const v8h*)(afrag + (size_t)a * 28672);
    const v8h* gb = (const v8h*)(bfrag + (size_t)b * 28672);
    for (int i = tid; i < 28672 / 8; i += 128) {
      ((v8h*)Af)[i] = ga[i];
      ((v8h*)Bf)[i] = gb[i];
    }
  }
#endif
  __syncthreads();

  float acc = 0.f;
  for (int pass = 0; pass < 7; ++pass) {
    const _Float16* Ap = Af + ((size_t)(pass * 4 + wave) * 2) * 512;
    v16h a0 = *(const v16h*)(Ap + lane * 16);
    v16h a1 = *(const v16h*)(Ap + 512 + lane * 16);

    float qmv[8];
#pragma unroll
    for (int r = 0; r < 8; ++r) {
      int p = pass * 64 + wave * 16 + hi_ * 8 + r;
      qmv[r] = (p < 441) ? qm[a * 441 + p] : 0.f;
    }

    float t0[8], t1[8], t2[8];
#pragma unroll
    for (int r = 0; r < 8; ++r) { t0[r] = t1[r] = t2[r] = -1e30f; }

    for (int qt = 0; qt < 28; ++qt) {
      const _Float16* Bp = Bf + (size_t)qt * 2 * 512;
      v8f cfr = {};
      cfr = wmma16(a0, *(const v16h*)(Bp + lane * 16), cfr);
      cfr = wmma16(a1, *(const v16h*)(Bp + 512 + lane * 16), cfr);
      int q = qt * 16 + col;
      bool valid = q < 441;
      float smv = valid ? sm[b * 441 + q] : 0.f;
#pragma unroll
      for (int r = 0; r < 8; ++r) {
        float v = valid ? cfr[r] * qmv[r] * smv : -1e30f;
        if (v > t2[r]) {
          if (v > t0[r])      { t2[r] = t1[r]; t1[r] = t0[r]; t0[r] = v; }
          else if (v > t1[r]) { t2[r] = t1[r]; t1[r] = v; }
          else                  t2[r] = v;
        }
      }
    }

#pragma unroll
    for (int r = 0; r < 8; ++r) {
      int rl = wave * 16 + hi_ * 8 + r;
      red[(rl * 16 + col) * 3 + 0] = t0[r];
      red[(rl * 16 + col) * 3 + 1] = t1[r];
      red[(rl * 16 + col) * 3 + 2] = t2[r];
    }
    __syncthreads();
    if (tid < 64) {
      float u0 = -1e30f, u1 = -1e30f, u2 = -1e30f;
      for (int l = 0; l < 48; ++l) {
        float v = red[tid * 48 + l];
        if (v > u2) {
          if (v > u0)      { u2 = u1; u1 = u0; u0 = v; }
          else if (v > u1) { u2 = u1; u1 = v; }
          else               u2 = v;
        }
      }
      acc += u0 + u1 + u2;
    }
    __syncthreads();
  }
  if (tid < 64) part[tid] = acc;
  __syncthreads();
  if (tid == 0) {
    float s = 0.f;
    for (int i = 0; i < 64; ++i) s += part[i];
    sim[blockIdx.x] = s;
  }
}

// Final 5-way classifier + softmax.
__global__ void fc_softmax_kernel(const float* __restrict__ sim,
                                  const float* __restrict__ w,
                                  const float* __restrict__ bias,
                                  float* __restrict__ out) {
  int a = blockIdx.x * blockDim.x + threadIdx.x;
  if (a >= 75) return;
  float lg[5]; float mx = -1e30f;
#pragma unroll
  for (int j = 0; j < 5; ++j) {
    float s = bias[j];
#pragma unroll
    for (int k = 0; k < 5; ++k) s += sim[a * 5 + k] * w[j * 5 + k];
    lg[j] = s; mx = fmaxf(mx, s);
  }
  float den = 0.f;
#pragma unroll
  for (int j = 0; j < 5; ++j) { lg[j] = __expf(lg[j] - mx); den += lg[j]; }
#pragma unroll
  for (int j = 0; j < 5; ++j) out[a * 5 + j] = lg[j] / den;
}

__global__ void cast_h_kernel(const float* __restrict__ in,
                              _Float16* __restrict__ out, int n) {
  int i = blockIdx.x * blockDim.x + threadIdx.x;
  if (i < n) out[i] = (_Float16)in[i];
}

// ---------------------------------------------------------------------------
// Host orchestration
// ---------------------------------------------------------------------------
extern "C" void kernel_launch(void* const* d_in, const int* in_sizes, int n_in,
                              void* d_out, int out_size, void* d_ws, size_t ws_size,
                              hipStream_t stream) {
  (void)in_sizes; (void)n_in; (void)out_size; (void)ws_size;
  const float* input1 = (const float*)d_in[0];
  const float* input2 = (const float*)d_in[1];
  const float *w1 = (const float*)d_in[2],  *g1 = (const float*)d_in[3],  *b1 = (const float*)d_in[4];
  const float *w2 = (const float*)d_in[5],  *g2 = (const float*)d_in[6],  *b2 = (const float*)d_in[7];
  const float *w3 = (const float*)d_in[8],  *g3 = (const float*)d_in[9],  *b3 = (const float*)d_in[10];
  const float *w4 = (const float*)d_in[11], *g4 = (const float*)d_in[12], *b4 = (const float*)d_in[13];
  const float *cw1 = (const float*)d_in[14], *cg1 = (const float*)d_in[15], *cb1 = (const float*)d_in[16];
  const float *cw2 = (const float*)d_in[17], *cg2 = (const float*)d_in[18], *cb2 = (const float*)d_in[19];
  const float *cw3 = (const float*)d_in[20], *cb3 = (const float*)d_in[21];
  const float *dw1 = (const float*)d_in[22], *dg1 = (const float*)d_in[23], *db1 = (const float*)d_in[24];
  const float *dw2 = (const float*)d_in[25], *dg2 = (const float*)d_in[26], *db2 = (const float*)d_in[27];
  const float *dw3 = (const float*)d_in[28], *db3 = (const float*)d_in[29];
  const float *sxw = (const float*)d_in[30], *sxb = (const float*)d_in[31];
  const float *syw = (const float*)d_in[32], *syb = (const float*)d_in[33];
  const float *tw  = (const float*)d_in[34], *tb  = (const float*)d_in[35];
  const float *fcw = (const float*)d_in[36], *fcb = (const float*)d_in[37];

  size_t off = 0;
  auto alloc = [&](size_t bytes) -> char* {
    off = (off + 255) & ~(size_t)255;
    char* p = (char*)d_ws + off;
    off += bytes;
    return p;
  };
  auto allocH = [&](size_t n) { return (_Float16*)alloc(n * sizeof(_Float16)); };
  auto allocF = [&](size_t n) { return (float*)alloc(n * sizeof(float)); };

  // Fragment-ordered f16 weight panels.
  _Float16 *w1f = allocH(2048);                  // KB=1 (K=27 padded to 32)
  _Float16 *w2f = allocH(36864), *w3f = allocH(36864), *w4f = allocH(36864);
  _Float16 *cw1f = allocH(36864), *cw2f = allocH(36864);

  _Float16* x84h  = allocH((size_t)75 * 3 * 84 * 84);   // f16 input cast
  _Float16* raw84 = allocH((size_t)75 * 64 * 84 * 84);
  _Float16* x42   = allocH((size_t)75 * 64 * 42 * 42);
  _Float16* raw42 = allocH((size_t)75 * 64 * 42 * 42);
  _Float16* x21   = allocH((size_t)75 * 64 * 441);
  _Float16* x21b  = allocH((size_t)75 * 64 * 441);
  _Float16* raw21 = allocH((size_t)75 * 64 * 441);
  _Float16* featq = allocH((size_t)75 * 64 * 441);
  _Float16* feats = allocH((size_t)5 * 64 * 441);
  _Float16* afragQ = allocH((size_t)75 * 28672);
  _Float16* bfragS = allocH((size_t)5 * 28672);
  _Float16* gpool = allocH((size_t)75 * 64 * 49);
  _Float16* g2raw = allocH((size_t)75 * 64 * 9);
  _Float16* g2b   = allocH((size_t)75 * 64 * 9);
  float*    vbuf  = allocF(75 * 64);
  _Float16* vh    = allocH(75 * 64);
  _Float16* d1raw = allocH((size_t)75 * 64 * 16);
  _Float16* d1    = allocH((size_t)75 * 64 * 16);
  _Float16* d2raw = allocH((size_t)75 * 64 * 100);
  _Float16* d2    = allocH((size_t)75 * 64 * 100);
  float*    logits = allocF(75 * 441);
  float*    mi    = allocF(75 * 2);
  float*    ginv  = allocF(75 * 4);
  float*    m1b   = allocF(75 * 441);
  float*    m2b   = allocF(5 * 441);
  float*    qmb   = allocF(75 * 441);
  float*    smb   = allocF(5 * 441);
  float*    scsh  = allocF(128);
  float*    simb  = allocF(375);

  auto conv_lds = [](int KB) { return (size_t)(4 * KB * 512 + KB * 4 * 512) * 2; };
  const size_t COS_LDS = 57344 + 57344 + 12288 + 256;   // ~124KB

  // Pre-swizzle conv weights into fragment order.
  prep_wfrag_kernel<<<1, 128, 0, stream>>>(w1, w1f, 1, 27);
  prep_wfrag_kernel<<<9, 256, 0, stream>>>(w2,  w2f,  18, 576);
  prep_wfrag_kernel<<<9, 256, 0, stream>>>(w3,  w3f,  18, 576);
  prep_wfrag_kernel<<<9, 256, 0, stream>>>(w4,  w4f,  18, 576);
  prep_wfrag_kernel<<<9, 256, 0, stream>>>(cw1, cw1f, 18, 576);
  prep_wfrag_kernel<<<9, 256, 0, stream>>>(cw2, cw2f, 18, 576);

  auto feature = [&](const float* inp, int B, _Float16* featOut) {
    // conv1 via WMMA: K = 27 padded to 32 (KB=1), f16 input.
    int nin = B * 3 * 84 * 84;
    cast_h_kernel<<<(nin + 255) / 256, 256, 0, stream>>>(inp, x84h, nin);
    dim3 gc1((84 * 84 + 63) / 64, B);
    conv_wmma_kernel<<<gc1, 128, conv_lds(1), stream>>>(x84h, w1f, raw84,
                                                        84, 84, 84, 84, 1, 1, 1, 3);
    bn_stats_kernel<<<64, 256, 0, stream>>>(raw84, g1, b1, scsh, B, 84 * 84);
    int tot42 = B * 64 * 42 * 42;
    bn_lrelu_pool_kernel<<<(tot42 + 255) / 256, 256, 0, stream>>>(raw84, scsh, x42, B, 84, 84, 2);

    dim3 gc2((42 * 42 + 63) / 64, B);
    conv_wmma_kernel<<<gc2, 128, conv_lds(18), stream>>>(x42, w2f, raw42,
                                                         42, 42, 42, 42, 1, 1, 18, 64);
    bn_stats_kernel<<<64, 256, 0, stream>>>(raw42, g2, b2, scsh, B, 42 * 42);
    int tot21 = B * 64 * 441;
    bn_lrelu_pool_kernel<<<(tot21 + 255) / 256, 256, 0, stream>>>(raw42, scsh, x21, B, 42, 42, 2);

    dim3 gc3((441 + 63) / 64, B);
    conv_wmma_kernel<<<gc3, 128, conv_lds(18), stream>>>(x21, w3f, raw21,
                                                         21, 21, 21, 21, 1, 1, 18, 64);
    bn_stats_kernel<<<64, 256, 0, stream>>>(raw21, g3, b3, scsh, B, 441);
    bn_lrelu_pool_kernel<<<(tot21 + 255) / 256, 256, 0, stream>>>(raw21, scsh, x21b, B, 21, 21, 1);

    conv_wmma_kernel<<<gc3, 128, conv_lds(18), stream>>>(x21b, w4f, raw21,
                                                         21, 21, 21, 21, 1, 1, 18, 64);
    bn_stats_kernel<<<64, 256, 0, stream>>>(raw21, g4, b4, scsh, B, 441);
    bn_lrelu_pool_kernel<<<(tot21 + 255) / 256, 256, 0, stream>>>(raw21, scsh, featOut, B, 21, 21, 1);
  };

  auto gaussian = [&](const _Float16* feat, int B, float* mOut) {
    dim3 gg((441 + 63) / 64, B);
    conv_wmma_kernel<<<gg, 128, conv_lds(18), stream>>>(feat, cw1f, raw21,
                                                        21, 21, 21, 21, 1, 1, 18, 64);
    bn_stats_kernel<<<64, 256, 0, stream>>>(raw21, cg1, cb1, scsh, B, 441);
    int totp = B * 64 * 49;
    bn_lrelu_pool_kernel<<<(totp + 255) / 256, 256, 0, stream>>>(raw21, scsh, gpool, B, 21, 21, 3);

    dim3 gg2(1, B);
    conv_wmma_kernel<<<gg2, 128, conv_lds(18), stream>>>(gpool, cw2f, g2raw,
                                                         7, 7, 3, 3, 2, 0, 18, 64);
    bn_stats_kernel<<<64, 256, 0, stream>>>(g2raw, cg2, cb2, scsh, B, 9);
    int tot9 = B * 64 * 9;
    bn_lrelu_pool_kernel<<<(tot9 + 255) / 256, 256, 0, stream>>>(g2raw, scsh, g2b, B, 3, 3, 1);

    cw3_kernel<<<(B * 64 + 255) / 256, 256, 0, stream>>>(g2b, cw3, cb3, vbuf, B);
    cast_h_kernel<<<(B * 64 + 255) / 256, 256, 0, stream>>>(vbuf, vh, B * 64);

    int td1 = B * 64 * 16;
    deconv_kernel<<<(td1 + 255) / 256, 256, 0, stream>>>(vh, dw1, nullptr, d1raw, nullptr,
                                                         B, 64, 64, 4, 1, 1, 4);
    bn_stats_kernel<<<64, 256, 0, stream>>>(d1raw, dg1, db1, scsh, B, 16);
    bn_lrelu_pool_kernel<<<(td1 + 255) / 256, 256, 0, stream>>>(d1raw, scsh, d1, B, 4, 4, 1);

    int td2 = B * 64 * 100;
    deconv_kernel<<<(td2 + 255) / 256, 256, 0, stream>>>(d1, dw2, nullptr, d2raw, nullptr,
                                                         B, 64, 64, 4, 2, 4, 10);
    bn_stats_kernel<<<64, 256, 0, stream>>>(d2raw, dg2, db2, scsh, B, 100);
    bn_lrelu_pool_kernel<<<(td2 + 255) / 256, 256, 0, stream>>>(d2raw, scsh, d2, B, 10, 10, 1);

    int td3 = B * 441;
    deconv_kernel<<<(td3 + 255) / 256, 256, 0, stream>>>(d2, dw3, db3, nullptr, logits,
                                                         B, 64, 1, 3, 2, 10, 21);
    mi_kernel<<<(B + 63) / 64, 64, 0, stream>>>(logits, mi, B);
    sigma_kernel<<<(B + 63) / 64, 64, 0, stream>>>(vbuf, sxw, sxb, syw, syb, tw, tb, ginv, B);
    mask_kernel<<<(B * 441 + 255) / 256, 256, 0, stream>>>(mi, ginv, mOut, B);
  };

  feature(input1, 75, featq);
  feature(input2, 5,  feats);
  gaussian(featq, 75, m1b);
  gaussian(feats, 5,  m2b);

  normmask_kernel<<<(75 * 441 + 255) / 256, 256, 0, stream>>>(featq, m1b, qmb, 75);
  normmask_kernel<<<(5 * 441 + 255) / 256, 256, 0, stream>>>(feats, m2b, smb, 5);

  prep_afrag_kernel<<<(75 * 1792 + 255) / 256, 256, 0, stream>>>(featq, afragQ, 75);
  prep_bfrag_kernel<<<(5 * 1792 + 255) / 256, 256, 0, stream>>>(feats, bfragS, 5);

  cossim_topk_kernel<<<375, 128, COS_LDS, stream>>>(afragQ, bfragS, qmb, smb, simb);
  fc_softmax_kernel<<<1, 128, 0, stream>>>(simb, fcw, fcb, (float*)d_out);
}
